// FasterSelfAttention_80049600463618
// MI455X (gfx1250) — matrix operand order repaired and checked by reading, not compiled, not run on hardware
//
#include <hip/hip_runtime.h>
#include <hip/hip_bf16.h>

// ---------------------------------------------------------------------------
// FasterSelfAttention for MI455X (gfx1250, wave32, WMMA bf16)
// B=2, S=2048, EMBED=1024, HEADS=16, DHEAD=64
// ---------------------------------------------------------------------------

#define BATCH   2
#define SEQ     2048
#define EMBED   1024
#define HEADS   16
#define DHEAD   64
#define NQKV    (3 * HEADS * DHEAD)   // 3072
#define MTOT    (BATCH * SEQ)         // 4096

typedef __bf16 v16bf __attribute__((ext_vector_type(16)));
typedef float  v8f   __attribute__((ext_vector_type(8)));

__device__ __forceinline__ unsigned short f32_to_bf16(float f) {
    unsigned int u = __float_as_uint(f);
    u = (u + 0x7FFFu + ((u >> 16) & 1u)) >> 16;   // round-to-nearest-even
    return (unsigned short)u;
}

__device__ __forceinline__ v16bf load_frag16(const unsigned short* p0,
                                             const unsigned short* p1) {
    union { v16bf v; uint4 q[2]; } u;
    u.q[0] = *reinterpret_cast<const uint4*>(p0);
    u.q[1] = *reinterpret_cast<const uint4*>(p1);
    return u.v;
}

__device__ __forceinline__ v8f wmma_bf16(v16bf a, v16bf b, v8f c) {
    // D(16x16,f32) = A(16x32,bf16) * B(32x16,bf16) + C
    return __builtin_amdgcn_wmma_f32_16x16x32_bf16(
        false, a, false, b, (short)0, c, false, false);
}

// ---------------------------------------------------------------------------
// Conversion kernels
// ---------------------------------------------------------------------------
// float4-vectorized f32 -> bf16 row-major copy (n must be multiple of 4)
__global__ void cvt_bf16_kernel(const float* __restrict__ in,
                                unsigned short* __restrict__ out, int n4) {
    int i = blockIdx.x * 256 + threadIdx.x;
    if (i < n4) {
        float4 f = reinterpret_cast<const float4*>(in)[i];
        union { unsigned short u[4]; uint2 v; } p;
        p.u[0] = f32_to_bf16(f.x);
        p.u[1] = f32_to_bf16(f.y);
        p.u[2] = f32_to_bf16(f.z);
        p.u[3] = f32_to_bf16(f.w);
        reinterpret_cast<uint2*>(out)[i] = p.v;
    }
}

// LDS-tiled transpose+convert: out[n][k] = bf16(in[k][n])
// in: K x N row-major f32; out: N x K row-major bf16. Block (32,8), 32x32 tile.
__global__ void __launch_bounds__(256)
cvt_bf16_T_kernel(const float* __restrict__ in,
                  unsigned short* __restrict__ out, int N, int K) {
    __shared__ float tile[32][33];
    const int n0 = blockIdx.x * 32;
    const int k0 = blockIdx.y * 32;
    const int tx = threadIdx.x;   // 0..31
    const int ty = threadIdx.y;   // 0..7
#pragma unroll
    for (int i = 0; i < 32; i += 8)
        tile[ty + i][tx] = in[(size_t)(k0 + ty + i) * N + n0 + tx];
    __syncthreads();
#pragma unroll
    for (int i = 0; i < 32; i += 8)
        out[(size_t)(n0 + ty + i) * K + k0 + tx] = f32_to_bf16(tile[tx][ty + i]);
}

// ---------------------------------------------------------------------------
// QKV GEMM: qkv = qs_bf16(4096x1024) @ WqkvT^T, scatter into
//   Q,K : [bh][s][d]   (bh = b*16+h)
//   V   : [bh][d][s]   (transposed via LDS so it is written with b128 stores)
// One wave: 32(M) x 64(N) tile, fully unrolled k-loop with two alternating
// fragment buffers (no register copies, address math folds into offsets).
// ---------------------------------------------------------------------------
__global__ void __launch_bounds__(128)
qkv_gemm_kernel(const unsigned short* __restrict__ A,    // 4096 x 1024 bf16
                const unsigned short* __restrict__ Bt,   // 3072 x 1024 bf16 (pre-transposed)
                unsigned short* __restrict__ Qb,
                unsigned short* __restrict__ Kb,
                unsigned short* __restrict__ Vt) {
    __shared__ __align__(16) unsigned short Vlds[4][64][40];  // per-wave V transpose tile

    const int lane = threadIdx.x & 31;
    const int w    = threadIdx.x >> 5;
    const int col  = lane & 15;
    const int half = lane >> 4;
    const int m0    = blockIdx.x * 32;
    const int nbase = (blockIdx.y * 4 + w) * 64;

    const unsigned short* arow0 = A + (size_t)(m0 + col) * EMBED;
    const unsigned short* arow1 = A + (size_t)(m0 + 16 + col) * EMBED;

    auto loadA = [&](int k0, v16bf* dst) {
        dst[0] = load_frag16(arow0 + k0 + half * 8, arow0 + k0 + 16 + half * 8);
        dst[1] = load_frag16(arow1 + k0 + half * 8, arow1 + k0 + 16 + half * 8);
    };
    auto loadB = [&](int k0, v16bf* dst) {
#pragma unroll
        for (int t = 0; t < 4; ++t) {
            const unsigned short* bp =
                Bt + (size_t)(nbase + t * 16 + col) * EMBED + k0 + half * 8;
            dst[t] = load_frag16(bp, bp + 16);
        }
    };

    v8f acc[2][4] = {};
    v16bf abuf[2][2], bbuf[2][4];
    loadA(0, abuf[0]);
    loadB(0, bbuf[0]);
#pragma unroll
    for (int it = 0; it < EMBED / 32; ++it) {
        const int cur = it & 1, nxt = cur ^ 1;
        const int k0 = it * 32;
        if (k0 + 32 < EMBED) {                // static after unroll
            loadA(k0 + 32, abuf[nxt]);
            loadB(k0 + 32, bbuf[nxt]);
        }
#pragma unroll
        for (int mt = 0; mt < 2; ++mt)
#pragma unroll
            for (int t = 0; t < 4; ++t)
                acc[mt][t] = wmma_bf16(abuf[cur][mt], bbuf[cur][t], acc[mt][t]);
    }

    const int head3 = nbase >> 6;         // 0..47
    const int which = head3 >> 4;         // 0=Q,1=K,2=V
    const int h     = head3 & 15;
    const int b_    = m0 >> 11;           // batch (m0..m0+31 stay in one batch)
    const int s0    = m0 & (SEQ - 1);
    const size_t bh = (size_t)(b_ * HEADS + h);

    if (which < 2) {
        unsigned short* dst = (which == 0) ? Qb : Kb;
#pragma unroll
        for (int mt = 0; mt < 2; ++mt)
#pragma unroll
            for (int t = 0; t < 4; ++t)
#pragma unroll
                for (int r = 0; r < 8; ++r) {
                    int s = s0 + mt * 16 + half * 8 + r;
                    int d = t * 16 + col;
                    dst[(bh * SEQ + s) * DHEAD + d] = f32_to_bf16(acc[mt][t][r]);
                }
    } else {
        // transpose 32(s) x 64(d) tile through LDS, store contiguous along s
#pragma unroll
        for (int mt = 0; mt < 2; ++mt)
#pragma unroll
            for (int t = 0; t < 4; ++t)
#pragma unroll
                for (int r = 0; r < 8; ++r)
                    Vlds[w][t * 16 + col][mt * 16 + half * 8 + r] =
                        f32_to_bf16(acc[mt][t][r]);
        __threadfence_block();            // order ds_store -> ds_load (same wave)
#pragma unroll
        for (int dd = 0; dd < 2; ++dd) {
            int d = dd * 32 + lane;
            const uint4* src = reinterpret_cast<const uint4*>(&Vlds[w][d][0]);
            uint4* dstp = reinterpret_cast<uint4*>(&Vt[(bh * DHEAD + d) * SEQ + s0]);
#pragma unroll
            for (int q = 0; q < 4; ++q) dstp[q] = src[q];
        }
    }
}

// ---------------------------------------------------------------------------
// Flash attention (causal), one wave per 16-row Q block, key blocks of 32.
// K/V fragments are double-buffered across key blocks so the next block's 8 fragment
// loads overlap the current block's softmax + PV WMMAs. Scores use base-2
// online softmax with __shfl_xor row reductions; P is transposed C->A layout
// through LDS; PV = 4 WMMA on resident V fragments.
// ---------------------------------------------------------------------------
__global__ void __launch_bounds__(128)
flash_attn_kernel(const unsigned short* __restrict__ Qb,
                  const unsigned short* __restrict__ Kb,
                  const unsigned short* __restrict__ Vt,
                  unsigned short* __restrict__ O) {
    __shared__ __align__(16) unsigned short Plds[4][16][40];

    const int lane = threadIdx.x & 31;
    const int w    = threadIdx.x >> 5;
    const int col  = lane & 15;
    const int half = lane >> 4;
    const int bh = blockIdx.x;                // 0..31
    const int q0 = (blockIdx.y * 4 + w) * 16; // q-block start row

    const float SCL2 = 0.125f * 1.4426950408889634f;  // (1/sqrt(64))*log2(e)

    const unsigned short* qrow = Qb + ((size_t)bh * SEQ + q0 + col) * DHEAD;
    v16bf qa0 = load_frag16(qrow + half * 8,      qrow + 16 + half * 8);
    v16bf qa1 = load_frag16(qrow + 32 + half * 8, qrow + 48 + half * 8);

    v8f acc[4] = {};
    float mrow[8], lrow[8];
#pragma unroll
    for (int r = 0; r < 8; ++r) { mrow[r] = -1e30f; lrow[r] = 0.0f; }

    auto loadKV = [&](int k0, v16bf* kf0, v16bf* kf1, v16bf* vbv) {
#pragma unroll
        for (int t = 0; t < 2; ++t) {
            const unsigned short* kp =
                Kb + ((size_t)bh * SEQ + k0 + t * 16 + col) * DHEAD + half * 8;
            kf0[t] = load_frag16(kp,      kp + 16);
            kf1[t] = load_frag16(kp + 32, kp + 48);
        }
#pragma unroll
        for (int n = 0; n < 4; ++n) {
            const unsigned short* vp =
                Vt + ((size_t)bh * DHEAD + n * 16 + col) * SEQ + k0 + half * 8;
            vbv[n] = load_frag16(vp, vp + 16);
        }
    };

    auto process = [&](int k0, const v16bf* kf0, const v16bf* kf1,
                       const v16bf* vbv) {
        v8f st[2];
#pragma unroll
        for (int t = 0; t < 2; ++t) {
            v8f z = {};
            z = wmma_bf16(qa0, kf0[t], z);
            z = wmma_bf16(qa1, kf1[t], z);
            st[t] = z;
        }
        float bmax[8];
#pragma unroll
        for (int r = 0; r < 8; ++r) {
            int qr = q0 + half * 8 + r;
            float v0 = (k0 + col      <= qr) ? st[0][r] * SCL2 : -1e9f;
            float v1 = (k0 + 16 + col <= qr) ? st[1][r] * SCL2 : -1e9f;
            st[0][r] = v0; st[1][r] = v1;
            float v = fmaxf(v0, v1);
            v = fmaxf(v, __shfl_xor(v, 1));
            v = fmaxf(v, __shfl_xor(v, 2));
            v = fmaxf(v, __shfl_xor(v, 4));
            v = fmaxf(v, __shfl_xor(v, 8));
            bmax[r] = v;
        }
#pragma unroll
        for (int r = 0; r < 8; ++r) {
            float mn  = fmaxf(mrow[r], bmax[r]);
            float scl = exp2f(mrow[r] - mn);
            mrow[r] = mn;
            float p0 = exp2f(st[0][r] - mn);
            float p1 = exp2f(st[1][r] - mn);
            float rs = p0 + p1;
            rs += __shfl_xor(rs, 1);
            rs += __shfl_xor(rs, 2);
            rs += __shfl_xor(rs, 4);
            rs += __shfl_xor(rs, 8);
            lrow[r] = lrow[r] * scl + rs;
#pragma unroll
            for (int n = 0; n < 4; ++n) acc[n][r] *= scl;
            Plds[w][half * 8 + r][col]      = f32_to_bf16(p0);
            Plds[w][half * 8 + r][16 + col] = f32_to_bf16(p1);
        }
        __threadfence_block();
        v16bf pa = load_frag16(&Plds[w][col][half * 8],
                               &Plds[w][col][16 + half * 8]);
#pragma unroll
        for (int n = 0; n < 4; ++n)
            acc[n] = wmma_bf16(pa, vbv[n], acc[n]);
    };

    const int nkb = q0 / 32 + 1;
    v16bf kf0A[2], kf1A[2], vbA[4];
    v16bf kf0B[2], kf1B[2], vbB[4];
    loadKV(0, kf0A, kf1A, vbA);
    int kb = 0;
    while (true) {
        if (kb + 1 < nkb) loadKV((kb + 1) * 32, kf0B, kf1B, vbB);
        process(kb * 32, kf0A, kf1A, vbA);
        if (++kb >= nkb) break;
        if (kb + 1 < nkb) loadKV((kb + 1) * 32, kf0A, kf1A, vbA);
        process(kb * 32, kf0B, kf1B, vbB);
        if (++kb >= nkb) break;
    }

    // ---- epilogue: normalize and write attn output bf16 [b][s][h*64+d]
    const int b = bh >> 4, h = bh & 15;
#pragma unroll
    for (int r = 0; r < 8; ++r) {
        float inv = 1.0f / lrow[r];
        size_t row = (size_t)b * SEQ + q0 + half * 8 + r;
#pragma unroll
        for (int n = 0; n < 4; ++n) {
            O[row * (HEADS * DHEAD) + h * DHEAD + n * 16 + col] =
                f32_to_bf16(acc[n][r] * inv);
        }
    }
}

// ---------------------------------------------------------------------------
// Output projection: out(4096x1024 f32) = attn_bf16 @ WoutT^T + bout
// Same fully-unrolled double-buffered 32x64-per-wave WMMA structure.
// ---------------------------------------------------------------------------
__global__ void __launch_bounds__(128)
out_proj_kernel(const unsigned short* __restrict__ A,    // 4096 x 1024 bf16
                const unsigned short* __restrict__ Bt,   // 1024 x 1024 bf16 (pre-transposed)
                const float* __restrict__ bias,
                float* __restrict__ out) {
    const int lane = threadIdx.x & 31;
    const int w    = threadIdx.x >> 5;
    const int col  = lane & 15;
    const int half = lane >> 4;
    const int m0    = blockIdx.x * 32;
    const int nbase = (blockIdx.y * 4 + w) * 64;

    const unsigned short* arow0 = A + (size_t)(m0 + col) * EMBED;
    const unsigned short* arow1 = A + (size_t)(m0 + 16 + col) * EMBED;

    auto loadA = [&](int k0, v16bf* dst) {
        dst[0] = load_frag16(arow0 + k0 + half * 8, arow0 + k0 + 16 + half * 8);
        dst[1] = load_frag16(arow1 + k0 + half * 8, arow1 + k0 + 16 + half * 8);
    };
    auto loadB = [&](int k0, v16bf* dst) {
#pragma unroll
        for (int t = 0; t < 4; ++t) {
            const unsigned short* bp =
                Bt + (size_t)(nbase + t * 16 + col) * EMBED + k0 + half * 8;
            dst[t] = load_frag16(bp, bp + 16);
        }
    };

    v8f acc[2][4] = {};
    v16bf abuf[2][2], bbuf[2][4];
    loadA(0, abuf[0]);
    loadB(0, bbuf[0]);
#pragma unroll
    for (int it = 0; it < EMBED / 32; ++it) {
        const int cur = it & 1, nxt = cur ^ 1;
        const int k0 = it * 32;
        if (k0 + 32 < EMBED) {
            loadA(k0 + 32, abuf[nxt]);
            loadB(k0 + 32, bbuf[nxt]);
        }
#pragma unroll
        for (int mt = 0; mt < 2; ++mt)
#pragma unroll
            for (int t = 0; t < 4; ++t)
                acc[mt][t] = wmma_bf16(abuf[cur][mt], bbuf[cur][t], acc[mt][t]);
    }

#pragma unroll
    for (int mt = 0; mt < 2; ++mt)
#pragma unroll
        for (int t = 0; t < 4; ++t)
#pragma unroll
            for (int r = 0; r < 8; ++r) {
                int m = m0 + mt * 16 + half * 8 + r;
                int n = nbase + t * 16 + col;
                out[(size_t)m * EMBED + n] = acc[mt][t][r] + bias[n];
            }
}

// ---------------------------------------------------------------------------
// Launch
// ---------------------------------------------------------------------------
extern "C" void kernel_launch(void* const* d_in, const int* in_sizes, int n_in,
                              void* d_out, int out_size, void* d_ws, size_t ws_size,
                              hipStream_t stream) {
    const float* qs   = (const float*)d_in[0];
    // d_in[1] = mask (causal, handled analytically)
    const float* Wqkv = (const float*)d_in[2];
    const float* Wout = (const float*)d_in[3];
    const float* bout = (const float*)d_in[4];
    float* out = (float*)d_out;

    char* ws = (char*)d_ws;
    unsigned short* qsb   = (unsigned short*)(ws);                      //  8 MiB
    unsigned short* wqkvT = (unsigned short*)(ws + (8u  << 20));        //  6 MiB
    unsigned short* woutT = (unsigned short*)(ws + (14u << 20));        //  2 MiB
    unsigned short* Qb    = (unsigned short*)(ws + (16u << 20));        //  8 MiB
    unsigned short* Kb    = (unsigned short*)(ws + (24u << 20));        //  8 MiB
    unsigned short* Vt    = (unsigned short*)(ws + (32u << 20));        //  8 MiB
    unsigned short* attnO = (unsigned short*)(ws + (40u << 20));        //  8 MiB

    // 1) bf16 staging (vectorized copy + LDS-tiled transposes)
    cvt_bf16_kernel<<<(MTOT * EMBED / 4) / 256, 256, 0, stream>>>(qs, qsb, MTOT * EMBED / 4);
    cvt_bf16_T_kernel<<<dim3(NQKV / 32, EMBED / 32), dim3(32, 8), 0, stream>>>(
        Wqkv, wqkvT, NQKV, EMBED);
    cvt_bf16_T_kernel<<<dim3(EMBED / 32, EMBED / 32), dim3(32, 8), 0, stream>>>(
        Wout, woutT, EMBED, EMBED);

    // 2) QKV projection (WMMA) with Q/K/V scatter (V transposed via LDS)
    qkv_gemm_kernel<<<dim3(MTOT / 32, NQKV / 256), 128, 0, stream>>>(qsb, wqkvT, Qb, Kb, Vt);

    // 3) causal flash attention (WMMA + online softmax, double-buffered K/V)
    flash_attn_kernel<<<dim3(BATCH * HEADS, SEQ / 64), 128, 0, stream>>>(Qb, Kb, Vt, attnO);

    // 4) output projection (WMMA) + bias, f32 out
    out_proj_kernel<<<dim3(MTOT / 32, EMBED / 256), 128, 0, stream>>>(attnO, woutT, bout, out);
}
